// MetaPathTokenGenerator_3289944949564
// MI455X (gfx1250) — matrix operand order repaired and checked by reading, not compiled, not run on hardware
//
#include <hip/hip_runtime.h>

#define N_NODES 10000
#define N_EDGES 320000
#define D_FEAT 128
#define OUT_DIM 128
#define SCALES 3
#define K1PAD 256            // concat(src,dest); gap column handled as rank-1 VALU update
#define K1TILES 8
#define K2TILES 4
#define NTILES 8
#define B1_FRAGS (SCALES * K1TILES * NTILES)
#define B2_FRAGS (SCALES * K2TILES * NTILES)
#define FRAG_USHORTS 512     // 32 lanes * 16 bf16 elements

typedef __attribute__((ext_vector_type(16))) __bf16        v16bf;
typedef __attribute__((ext_vector_type(8)))  float         v8f;
typedef __attribute__((ext_vector_type(4)))  float         v4f;
typedef __attribute__((ext_vector_type(4)))  unsigned int  v4u;
typedef __attribute__((ext_vector_type(2)))  unsigned int  v2u;

__device__ __forceinline__ unsigned short f2bf(float f) {
  union { float f; unsigned int u; } c; c.f = f;
  unsigned int r = c.u + 0x7FFFu + ((c.u >> 16) & 1u);
  return (unsigned short)(r >> 16);
}

// ---------------------------------------------------------------------------
// Prep: convert W1/W2 (f32, row-major [S][K][N]) into bf16 WMMA B-fragments.
// Fragment layout (16-bit B 32x16, this code's contiguous-k order, not the ISA table order): lane l -> column n = l&15,
// K-half = (l>>4)*16; 16 consecutive K values per lane, stored contiguously
// so the GEMM kernel loads each fragment as two b128 loads per lane.
// W1 row 256 (temporal gap) is NOT converted: it is applied in f32 as a
// rank-1 update in the main kernel.
// ---------------------------------------------------------------------------
__global__ void prep_weights(const float* __restrict__ W1,
                             const float* __restrict__ W2,
                             unsigned short* __restrict__ wsB1,
                             unsigned short* __restrict__ wsB2) {
  int tid = blockIdx.x * blockDim.x + threadIdx.x;
  const int total1 = B1_FRAGS * FRAG_USHORTS;
  const int total2 = B2_FRAGS * FRAG_USHORTS;
  if (tid < total1) {
    int frag = tid >> 9, r = tid & 511;
    int lane = r >> 4, j = r & 15;
    int nt = frag & 7, kb = (frag >> 3) % K1TILES, s = frag / (K1TILES * NTILES);
    int n = nt * 16 + (lane & 15);
    int k = kb * 32 + ((lane >> 4) * 16) + j;            // k in [0,256)
    wsB1[tid] = f2bf(W1[((size_t)s * 257 + k) * OUT_DIM + n]);
  } else if (tid < total1 + total2) {
    int t = tid - total1;
    int frag = t >> 9, r = t & 511;
    int lane = r >> 4, j = r & 15;
    int nt = frag & 7, kb = (frag >> 3) & 3, s = frag / (K2TILES * NTILES);
    int n = nt * 16 + (lane & 15);
    int k = kb * 32 + ((lane >> 4) * 16) + j;
    wsB2[t] = f2bf(W2[((size_t)s * OUT_DIM + k) * OUT_DIM + n]);
  }
}

// ---------------------------------------------------------------------------
// Main kernel: 4 waves per block, each wave owns one 16-edge tile and the
// full 128 output columns. bf16 WMMA (16x16x32) with f32 accumulation.
// ---------------------------------------------------------------------------
__launch_bounds__(128, 1)
__global__ void edge_mlp(const float* __restrict__ nf,
                         const float* __restrict__ temporal,
                         const float* __restrict__ W1,
                         const float* __restrict__ b1,
                         const float* __restrict__ b2,
                         const int* __restrict__ eidx,
                         const unsigned short* __restrict__ wsB1,
                         const unsigned short* __restrict__ wsB2,
                         float* __restrict__ out) {
  __shared__ unsigned short ef[4][16][K1PAD];   // edge features, bf16, 32 KB
  __shared__ unsigned short hb[4][16][OUT_DIM]; // layer-1 activations, 16 KB
  __shared__ float gaps[4][16];                 // temporal gaps (f32), 256 B

  const int lane = threadIdx.x & 31;
  const int wv = threadIdx.x >> 5;
  const int tile = blockIdx.x * 4 + wv;
  const int e0 = tile * 16;
  const int* __restrict__ srcI = eidx;
  const int* __restrict__ dstI = eidx + N_EDGES;

  // ---- stage 16 edges' features into LDS (coalesced: 32 lanes per row) ----
  for (int m = 0; m < 16; ++m) {
    const int e = e0 + m;
    const int sN = srcI[e], dN = dstI[e];
    v4f a = *(const v4f*)(nf + (size_t)sN * D_FEAT + lane * 4);
    v4f b = *(const v4f*)(nf + (size_t)dN * D_FEAT + lane * 4);
    v2u pa, pb;
    pa.x = (unsigned)f2bf(a.x) | ((unsigned)f2bf(a.y) << 16);
    pa.y = (unsigned)f2bf(a.z) | ((unsigned)f2bf(a.w) << 16);
    pb.x = (unsigned)f2bf(b.x) | ((unsigned)f2bf(b.y) << 16);
    pb.y = (unsigned)f2bf(b.z) | ((unsigned)f2bf(b.w) << 16);
    *(v2u*)&ef[wv][m][lane * 4] = pa;
    *(v2u*)&ef[wv][m][D_FEAT + lane * 4] = pb;
  }
  if (lane < 16) {  // temporal gap, kept in f32 for the rank-1 update
    const int e = e0 + lane;
    gaps[wv][lane] = temporal[srcI[e]] - temporal[dstI[e]];
  }
  asm volatile("s_wait_dscnt 0" ::: "memory");

  // A-fragment addressing: row m = lane&15,
  // u[0..15] = K + khalf .. +15 (khalf = (lane>>4)*16), matching B fragments.
  const int mrow = lane & 15;
  const int khalfA = (lane >> 4) * 16;
  const int laneOffB = lane * 16;
  const int mbase = (lane < 16) ? 0 : 8;
  const int ncol = lane & 15;

  // gap values for the 8 rows this lane's acc elements cover
  v4f gap_lo = *(const v4f*)&gaps[wv][mbase];
  v4f gap_hi = *(const v4f*)&gaps[wv][mbase + 4];

  v8f acc2[NTILES];
#pragma unroll
  for (int nt = 0; nt < NTILES; ++nt) acc2[nt] = v8f{};

  for (int s = 0; s < SCALES; ++s) {
    // ---------------- layer 1: [16 x 256] x [256 x 128] ----------------
    v8f acc1[NTILES];
#pragma unroll
    for (int nt = 0; nt < NTILES; ++nt) acc1[nt] = v8f{};

    for (int kb = 0; kb < K1TILES; ++kb) {
      union { v16bf v; v4u q[2]; } ua;
      const int k0 = kb * 32 + khalfA;
      ua.q[0] = *(const v4u*)&ef[wv][mrow][k0];
      ua.q[1] = *(const v4u*)&ef[wv][mrow][k0 + 8];
      const unsigned short* fb = wsB1 + (size_t)((s * K1TILES + kb) * NTILES) * FRAG_USHORTS;
#pragma unroll
      for (int nt = 0; nt < NTILES; ++nt) {
        union { v16bf v; v4u q[2]; } ub;
        const unsigned short* fp = fb + nt * FRAG_USHORTS + laneOffB;
        ub.q[0] = *(const v4u*)(fp);
        ub.q[1] = *(const v4u*)(fp + 8);
        acc1[nt] = __builtin_amdgcn_wmma_f32_16x16x32_bf16(
            false, ua.v, false, ub.v, (short)0, acc1[nt], false, false);
      }
    }

    // rank-1 gap update (f32) + bias + ReLU, write h to LDS as bf16
    {
      const float* w256 = W1 + ((size_t)s * 257 + 256) * OUT_DIM;
#pragma unroll
      for (int nt = 0; nt < NTILES; ++nt) {
        const int col = nt * 16 + ncol;
        const float bv = b1[s * OUT_DIM + col];
        const float wg = w256[col];
#pragma unroll
        for (int v = 0; v < 8; ++v) {
          const float g = (v < 4) ? gap_lo[v & 3] : gap_hi[v & 3];
          float h = acc1[nt][v] + bv + g * wg;
          h = h > 0.0f ? h : 0.0f;
          hb[wv][mbase + v][col] = f2bf(h);
        }
      }
    }
    asm volatile("s_wait_dscnt 0" ::: "memory");

    // ---------------- layer 2: [16 x 128] x [128 x 128], accumulate scales
    for (int kb = 0; kb < K2TILES; ++kb) {
      union { v16bf v; v4u q[2]; } ua;
      const int k0 = kb * 32 + khalfA;
      ua.q[0] = *(const v4u*)&hb[wv][mrow][k0];
      ua.q[1] = *(const v4u*)&hb[wv][mrow][k0 + 8];
      const unsigned short* fb = wsB2 + (size_t)((s * K2TILES + kb) * NTILES) * FRAG_USHORTS;
#pragma unroll
      for (int nt = 0; nt < NTILES; ++nt) {
        union { v16bf v; v4u q[2]; } ub;
        const unsigned short* fp = fb + nt * FRAG_USHORTS + laneOffB;
        ub.q[0] = *(const v4u*)(fp);
        ub.q[1] = *(const v4u*)(fp + 8);
        acc2[nt] = __builtin_amdgcn_wmma_f32_16x16x32_bf16(
            false, ua.v, false, ub.v, (short)0, acc2[nt], false, false);
      }
    }
  }

  // ---- epilogue: mean over scales + mean bias; half-wave coalesced stores
  const float inv3 = 1.0f / 3.0f;
#pragma unroll
  for (int nt = 0; nt < NTILES; ++nt) {
    const int col = nt * 16 + ncol;
    const float bm = (b2[col] + b2[OUT_DIM + col] + b2[2 * OUT_DIM + col]) * inv3;
#pragma unroll
    for (int v = 0; v < 8; ++v) {
      out[(size_t)(e0 + mbase + v) * OUT_DIM + col] = acc2[nt][v] * inv3 + bm;
    }
  }
}

extern "C" void kernel_launch(void* const* d_in, const int* in_sizes, int n_in,
                              void* d_out, int out_size, void* d_ws, size_t ws_size,
                              hipStream_t stream) {
  const float* nf       = (const float*)d_in[0];   // [10000,128] f32
  const float* temporal = (const float*)d_in[1];   // [10000] f32
  const float* W1       = (const float*)d_in[2];   // [3,257,128] f32
  const float* b1       = (const float*)d_in[3];   // [3,128] f32
  const float* W2       = (const float*)d_in[4];   // [3,128,128] f32
  const float* b2       = (const float*)d_in[5];   // [3,128] f32
  const int*   eidx     = (const int*)d_in[6];     // [2,320000]
  float* out = (float*)d_out;

  // workspace: pre-swizzled bf16 B-fragments (~288 KB)
  unsigned short* wsB1 = (unsigned short*)d_ws;
  unsigned short* wsB2 = wsB1 + (size_t)B1_FRAGS * FRAG_USHORTS;

  const int prepTotal = (B1_FRAGS + B2_FRAGS) * FRAG_USHORTS;
  prep_weights<<<(prepTotal + 255) / 256, 256, 0, stream>>>(W1, W2, wsB1, wsB2);

  const int nTiles = N_EDGES / 16;            // 20000 edge tiles
  edge_mlp<<<nTiles / 4, 128, 0, stream>>>(nf, temporal, W1, b1, b2, eidx,
                                           wsB1, wsB2, out);
}
